// WindowAttention3D_51479478010210
// MI455X (gfx1250) — matrix operand order repaired and checked by reading, not compiled, not run on hardware
//
#include <hip/hip_runtime.h>
#include <hip/hip_bf16.h>

typedef __attribute__((ext_vector_type(16))) _Float16 v16h;
typedef __attribute__((ext_vector_type(8)))  float    v8f;

#define NTOK 512
#define DIM 192
#define HEADS 6
#define HD 32
#define BW 64
#define QKVN 576           // 3*DIM
#define NROWS 32768        // BW*NTOK
#define TAB_ROWS 3375      // 15*15*15
#define WAVES 4            // waves per attention block

// ---------------------------------------------------------------------------
// Kernel 1: CPB MLP  (3375 x 3) -> relu(@w1^T + b1) -> @w2^T  => tab6 (3375 x 6)
// ---------------------------------------------------------------------------
__global__ __launch_bounds__(256)
void cpb_mlp_kernel(const float* __restrict__ table,   // [3375][3]
                    const float* __restrict__ w1,      // [512][3]
                    const float* __restrict__ b1,      // [512]
                    const float* __restrict__ w2,      // [6][512]
                    float* __restrict__ tab6)          // [3375][6]
{
    int i = blockIdx.x;
    int t = threadIdx.x;
    float t0 = table[i * 3 + 0];
    float t1 = table[i * 3 + 1];
    float t2 = table[i * 3 + 2];
    float acc[HEADS] = {0.f, 0.f, 0.f, 0.f, 0.f, 0.f};
    for (int j = t; j < 512; j += 256) {
        float h = fmaf(t0, w1[j * 3 + 0],
                  fmaf(t1, w1[j * 3 + 1],
                  fmaf(t2, w1[j * 3 + 2], b1[j])));
        h = fmaxf(h, 0.0f);
#pragma unroll
        for (int hh = 0; hh < HEADS; ++hh) acc[hh] += h * w2[hh * 512 + j];
    }
    __shared__ float red[256];
#pragma unroll
    for (int hh = 0; hh < HEADS; ++hh) {
        red[t] = acc[hh];
        __syncthreads();
        for (int s = 128; s > 0; s >>= 1) {
            if (t < s) red[t] += red[t + s];
            __syncthreads();
        }
        if (t == 0) tab6[i * HEADS + hh] = red[0];
        __syncthreads();
    }
}

// ---------------------------------------------------------------------------
// Kernel 2: gather bias via rel_pos_index, apply 16*sigmoid  => bias (6,512,512)
// ---------------------------------------------------------------------------
__global__ __launch_bounds__(256)
void bias_gather_kernel(const int* __restrict__ rpi,     // [512*512]
                        const float* __restrict__ tab6,  // [3375][6]
                        float* __restrict__ bias)        // [6][512][512]
{
    int idx = blockIdx.x * 256 + threadIdx.x;
    if (idx >= NTOK * NTOK) return;
    int t = rpi[idx];
#pragma unroll
    for (int hh = 0; hh < HEADS; ++hh) {
        float v = tab6[t * HEADS + hh];
        bias[(size_t)hh * (NTOK * NTOK) + idx] = 16.0f / (1.0f + __expf(-v));
    }
}

// ---------------------------------------------------------------------------
// Kernel 3: QKV projection, WMMA f16 (f32 accumulate), store qkv as f16
// ---------------------------------------------------------------------------
__global__ __launch_bounds__(32)
void qkv_proj_kernel(const float* __restrict__ x,    // [32768][192]
                     const float* __restrict__ w,    // [576][192]
                     const float* __restrict__ qb,   // [192]
                     const float* __restrict__ vb,   // [192]
                     _Float16* __restrict__ qkv)     // [32768][576] (f16)
{
    int nt = blockIdx.x;                 // 0..35
    int mt = blockIdx.y;                 // 0..2047
    int l  = threadIdx.x;
    int lc = l & 15;
    int hi = l >> 4;                     // 0 or 1
    int m0 = mt * 16, n0 = nt * 16;

    v8f c = {};
#pragma unroll
    for (int kt = 0; kt < 6; ++kt) {
        int k0 = kt * 32;
        v16h a;
        const float* xr = x + (size_t)(m0 + lc) * DIM + k0;
#pragma unroll
        for (int e = 0; e < 16; ++e) {
            int v = e >> 1;
            int k = (hi << 4) + e + (v & 0);
            a[e] = (_Float16)xr[k];
        }
        v16h bf;
        const float* wr = w + (size_t)(n0 + lc) * DIM + k0 + (hi << 4);
#pragma unroll
        for (int e = 0; e < 16; ++e) bf[e] = (_Float16)wr[e];
        c = __builtin_amdgcn_wmma_f32_16x16x32_f16(false, a, false, bf,
                                                   (short)0, c, false, false);
    }
    int n = n0 + lc;
    float bias = (n < DIM) ? qb[n] : ((n >= 2 * DIM) ? vb[n - 2 * DIM] : 0.0f);
#pragma unroll
    for (int i = 0; i < 8; ++i) {
        int row = m0 + i + (hi << 3);
        qkv[(size_t)row * QKVN + n] = (_Float16)(c[i] + bias);
    }
}

// ---------------------------------------------------------------------------
// Kernel 4: flash attention. Block = 4 waves = one (b, h, 64-query-row group).
//   V (512x32 f16) async-staged to LDS once per block; online softmax in
//   registers with wave32 shuffles; WMMA for q@k^T and p@v.
// ---------------------------------------------------------------------------
__global__ __launch_bounds__(32 * WAVES)
void attn_kernel(const _Float16* __restrict__ qkv,  // [32768][576] f16
                 const float* __restrict__ mask,    // [64][512][512]
                 const float* __restrict__ bias,    // [6][512][512]
                 float* __restrict__ out)           // [64][6][512][32]
{
    __shared__ _Float16 vbuf[NTOK * HD];            // 32 KB: V row-major
    __shared__ float    ptile[WAVES][16 * 32];      // 8 KB: per-wave prob tile

    int tid  = threadIdx.x;
    int wave = tid >> 5;
    int l    = tid & 31;
    int lc   = l & 15;
    int hi   = l >> 4;                              // 0 or 1
    int h    = blockIdx.y;
    int b    = blockIdx.z;
    int q0   = (blockIdx.x * WAVES + wave) * 16;
    const float scale = 0.17677669529663689f;       // 1/sqrt(32)

    // --- async-stage V(512x32) into LDS (CDNA5 async-to-LDS path) ----------
    {
        const _Float16* vsrc = qkv + (size_t)(b * NTOK) * QKVN + 2 * DIM + h * HD;
        // 16-byte chunks: 4 per row, 2048 total, 128 threads -> 16 iterations
        for (int cch = tid; cch < NTOK * 4; cch += 32 * WAVES) {
            int row = cch >> 2;
            int seg = cch & 3;
            unsigned lds_addr = (unsigned)(uintptr_t)(&vbuf[row * HD + seg * 8]);
            unsigned long long gaddr =
                (unsigned long long)(uintptr_t)(vsrc + (size_t)row * QKVN + seg * 8);
            asm volatile("global_load_async_to_lds_b128 %0, %1, off"
                         :: "v"(lds_addr), "v"(gaddr) : "memory");
        }
        asm volatile("s_wait_asynccnt 0x0" ::: "memory");
    }
    __syncthreads();

    // --- Q fragment (A: 16x32, K = head_dim = 32, loaded once) -------------
    const _Float16* qbase = qkv + (size_t)(b * NTOK + q0) * QKVN + h * HD;
    v16h aq;
#pragma unroll
    for (int e = 0; e < 16; ++e) {
        int v = e >> 1;
        int k = (hi << 4) + e + (v & 0);
        aq[e] = qbase[(size_t)lc * QKVN + k];
    }

    const _Float16* kmat = qkv + (size_t)(b * NTOK) * QKVN + DIM + h * HD;
    const float* mrow = mask + ((size_t)b * NTOK + q0) * NTOK;
    const float* brow = bias + ((size_t)h * NTOK + q0) * NTOK;

    v8f o0 = {}, o1 = {};
    float mrun[8], lrun[8];
#pragma unroll
    for (int i = 0; i < 8; ++i) { mrun[i] = -3.0e38f; lrun[i] = 0.0f; }

    // --- flash loop: 16 steps x 32 keys ------------------------------------
    for (int kt = 0; kt < 16; ++kt) {
        int kk = kt * 32;

        // two score tiles: B[d][m] = K[kk+m][d] (lane = key column)
        v16h bk0, bk1;
        const _Float16* kr0 = kmat + (size_t)(kk + lc) * QKVN + (hi << 4);
        const _Float16* kr1 = kmat + (size_t)(kk + 16 + lc) * QKVN + (hi << 4);
#pragma unroll
        for (int e = 0; e < 16; ++e) { bk0[e] = kr0[e]; bk1[e] = kr1[e]; }
        v8f c0 = {}, c1 = {};
        c0 = __builtin_amdgcn_wmma_f32_16x16x32_f16(false, aq, false, bk0,
                                                    (short)0, c0, false, false);
        c1 = __builtin_amdgcn_wmma_f32_16x16x32_f16(false, aq, false, bk1,
                                                    (short)0, c1, false, false);

        // epilogue + online softmax (rows live in 16-lane halves)
#pragma unroll
        for (int i = 0; i < 8; ++i) {
            int row  = i + (hi << 3);
            int col0 = kk + lc;
            float s0 = c0[i] * scale + mrow[(size_t)row * NTOK + col0]
                                     + brow[(size_t)row * NTOK + col0];
            float s1 = c1[i] * scale + mrow[(size_t)row * NTOK + col0 + 16]
                                     + brow[(size_t)row * NTOK + col0 + 16];
            float cm = fmaxf(s0, s1);
#pragma unroll
            for (int off = 8; off > 0; off >>= 1) cm = fmaxf(cm, __shfl_xor(cm, off, 32));
            float mnew = fmaxf(mrun[i], cm);
            float fac  = __expf(mrun[i] - mnew);
            float p0   = __expf(s0 - mnew);
            float p1   = __expf(s1 - mnew);
            float rs   = p0 + p1;
#pragma unroll
            for (int off = 8; off > 0; off >>= 1) rs += __shfl_xor(rs, off, 32);
            lrun[i] = lrun[i] * fac + rs;
            mrun[i] = mnew;
            o0[i] *= fac;
            o1[i] *= fac;
            ptile[wave][row * 32 + lc]      = p0;
            ptile[wave][row * 32 + 16 + lc] = p1;
        }

        // A fragment from prob tile (f32 LDS -> f16)
        v16h ap;
#pragma unroll
        for (int e = 0; e < 16; ++e) {
            int v = e >> 1;
            int k = (hi << 4) + e + (v & 0);
            ap[e] = (_Float16)ptile[wave][lc * 32 + k];
        }
        // B fragments from LDS-staged V: B[k][n] = v[kk+k][n]
        v16h bv0, bv1;
#pragma unroll
        for (int e = 0; e < 16; ++e) {
            int k = kk + (hi << 4) + e;
            bv0[e] = vbuf[k * HD + lc];
            bv1[e] = vbuf[k * HD + 16 + lc];
        }
        o0 = __builtin_amdgcn_wmma_f32_16x16x32_f16(false, ap, false, bv0,
                                                    (short)0, o0, false, false);
        o1 = __builtin_amdgcn_wmma_f32_16x16x32_f16(false, ap, false, bv1,
                                                    (short)0, o1, false, false);
    }

    // --- finalize: divide by running sum, store f32 ------------------------
    float* ob = out + (((size_t)b * HEADS + h) * NTOK + q0) * HD;
#pragma unroll
    for (int i = 0; i < 8; ++i) {
        int row = i + (hi << 3);
        float inv = 1.0f / lrun[i];
        ob[(size_t)row * HD + lc]      = o0[i] * inv;
        ob[(size_t)row * HD + 16 + lc] = o1[i] * inv;
    }
}

// ---------------------------------------------------------------------------
extern "C" void kernel_launch(void* const* d_in, const int* in_sizes, int n_in,
                              void* d_out, int out_size, void* d_ws, size_t ws_size,
                              hipStream_t stream) {
    const float* x    = (const float*)d_in[0];   // (64,512,192)
    const float* mask = (const float*)d_in[1];   // (64,512,512)
    const float* wqkv = (const float*)d_in[2];   // (576,192)
    const float* qb   = (const float*)d_in[3];   // (192)
    const float* vb   = (const float*)d_in[4];   // (192)
    const float* w1   = (const float*)d_in[5];   // (512,3)
    const float* b1   = (const float*)d_in[6];   // (512)
    const float* w2   = (const float*)d_in[7];   // (6,512)
    const float* tab  = (const float*)d_in[8];   // (1,15,15,15,3) -> 3375x3
    const int*   rpi  = (const int*)d_in[9];     // (512,512)
    float* out = (float*)d_out;

    // workspace partition
    char* ws = (char*)d_ws;
    float*    tab6   = (float*)ws;                               // 81000 B
    float*    biasb  = (float*)(ws + 81920);                     // 6291456 B
    _Float16* qkv16  = (_Float16*)(ws + 81920 + 6291456);        // 37748736 B

    cpb_mlp_kernel<<<TAB_ROWS, 256, 0, stream>>>(tab, w1, b1, w2, tab6);
    bias_gather_kernel<<<(NTOK * NTOK + 255) / 256, 256, 0, stream>>>(rpi, tab6, biasb);
    qkv_proj_kernel<<<dim3(QKVN / 16, NROWS / 16), 32, 0, stream>>>(x, wqkv, qb, vb, qkv16);
    attn_kernel<<<dim3(NTOK / 16 / WAVES, HEADS, BW), 32 * WAVES, 0, stream>>>(
        qkv16, mask, biasb, out);
}
